// SelfAttention_24206435680278
// MI455X (gfx1250) — matrix operand order repaired and checked by reading, not compiled, not run on hardware
//
#include <hip/hip_runtime.h>

#define D_MODEL 1024
#define N_HEADS 16
#define D_K 64
#define BATCH 4
#define SEQ 2048
#define M_TOTAL (BATCH * SEQ) // 8192 rows

typedef __attribute__((ext_vector_type(16))) __bf16 v16bf;
typedef __attribute__((ext_vector_type(8)))  __bf16 bf16x8;
typedef __attribute__((ext_vector_type(8)))  float  v8f;
typedef __attribute__((ext_vector_type(4)))  float  v4f;
typedef __attribute__((ext_vector_type(4)))  int    v4i;

#define AS1 __attribute__((address_space(1)))
#define AS3 __attribute__((address_space(3)))

#if defined(__gfx1250__) && __has_builtin(__builtin_amdgcn_global_load_async_to_lds_b128)
#define USE_ASYNC_LDS 1
#else
#define USE_ASYNC_LDS 0
#endif

// ---------- helpers ----------

static __device__ __forceinline__ __bf16 f2bf(float f) {
  unsigned u = __builtin_bit_cast(unsigned, f);
  u += 0x7FFFu + ((u >> 16) & 1u);            // round-to-nearest-even
  unsigned short h = (unsigned short)(u >> 16);
  return __builtin_bit_cast(__bf16, h);
}

static __device__ __forceinline__ void wait_async0() {
#if defined(__gfx1250__)
#if __has_builtin(__builtin_amdgcn_s_wait_asynccnt)
  __builtin_amdgcn_s_wait_asynccnt(0);
#else
  asm volatile("s_wait_asynccnt 0x0" ::: "memory");
#endif
#endif
}

// A fragment (16x32, MxK) per ISA: lanes 0-15 row m=lane, K {0..7,16..23};
// lanes 16-31 row m=lane-16, K {8..15,24..31}.
static __device__ __forceinline__ v16bf load_fragA(const __bf16* base, int ld) {
  const int lane = threadIdx.x & 31;
  const __bf16* p = base + (lane & 15) * ld + ((lane >> 4) << 3);
  bf16x8 lo = *(const bf16x8*)(p);
  bf16x8 hi = *(const bf16x8*)(p + 16);
  v16bf f;
#pragma unroll
  for (int j = 0; j < 8; ++j) { f[j] = lo[j]; f[j + 8] = hi[j]; }
  return f;
}

// B fragment (32x16, KxN) from B^T stored row-major [N, ld] (torch [out,in]):
// lanes 0-15 col n=lane K {0..7,16..23}, lanes 16-31 col n=lane-16 K {8..15,24..31}.
static __device__ __forceinline__ v16bf load_fragB(const __bf16* base, int ld) {
  const int lane = threadIdx.x & 31;
  const __bf16* p = base + (lane & 15) * ld + ((lane >> 4) << 3);
  bf16x8 lo = *(const bf16x8*)(p);
  bf16x8 hi = *(const bf16x8*)(p + 16);
  v16bf f;
#pragma unroll
  for (int j = 0; j < 8; ++j) { f[j] = lo[j]; f[j + 8] = hi[j]; }
  return f;
}

// ---------- fp32 -> bf16 conversion (8-wide: 2x b128 in, 1x b128 out) ----------

__global__ __launch_bounds__(256)
void cvt_f32_bf16(const float* __restrict__ in, __bf16* __restrict__ out, int nchunks) {
  int c = blockIdx.x * blockDim.x + threadIdx.x;
  const int stride = gridDim.x * blockDim.x;
  for (; c < nchunks; c += stride) {
    v4f a = *(const v4f*)(in + (size_t)c * 8);
    v4f b = *(const v4f*)(in + (size_t)c * 8 + 4);
    bf16x8 o;
#pragma unroll
    for (int j = 0; j < 4; ++j) { o[j] = f2bf(a[j]); o[j + 4] = f2bf(b[j]); }
    *(bf16x8*)(out + (size_t)c * 8) = o;
  }
}

// ---------- GEMM: out = A[M,1024] @ W[1024,1024]^T + bias ----------
// One wave computes a 32(M) x 64(N) tile: per 32-wide k-step, 2 A-frags
// shared across 4 B-frags -> 8 WMMAs per 12 b128 loads.

template <bool OUT_F32>
__global__ __launch_bounds__(256)
void gemm_xwT(const __bf16* __restrict__ A, const __bf16* __restrict__ W,
              const float* __restrict__ bias, void* __restrict__ outp, int M) {
  const int K = D_MODEL, N = D_MODEL;
  const int wid = blockIdx.x * (blockDim.x >> 5) + (threadIdx.x >> 5);
  const int NT = N / 64; // 16
  const int mt = wid / NT, nt = wid % NT;
  if (mt * 32 >= M) return; // wave-uniform
  const int row0 = mt * 32, n0 = nt * 64;
  const int lane = threadIdx.x & 31;

  const v8f zero8 = {0.f, 0.f, 0.f, 0.f, 0.f, 0.f, 0.f, 0.f};
  v8f acc[2][4] = {{zero8, zero8, zero8, zero8}, {zero8, zero8, zero8, zero8}};

  for (int k = 0; k < K; k += 32) {
    // prefetch next k-step operand lines (lowers to global_prefetch_b8)
    __builtin_prefetch(A + (size_t)(row0 + (threadIdx.x & 31)) * K + k + 32, 0, 3);
    __builtin_prefetch(W + (size_t)(n0 + (threadIdx.x & 15)) * K + k + 32, 0, 3);

    const v16bf a0 = load_fragA(A + (size_t)row0 * K + k, K);
    const v16bf a1 = load_fragA(A + (size_t)(row0 + 16) * K + k, K);
#pragma unroll
    for (int ns = 0; ns < 4; ++ns) {
      const v16bf b = load_fragB(W + (size_t)(n0 + ns * 16) * K + k, K);
      acc[0][ns] = __builtin_amdgcn_wmma_f32_16x16x32_bf16(
          false, a0, false, b, (short)0, acc[0][ns], false, false);
      acc[1][ns] = __builtin_amdgcn_wmma_f32_16x16x32_bf16(
          false, a1, false, b, (short)0, acc[1][ns], false, false);
    }
  }

#pragma unroll
  for (int ms = 0; ms < 2; ++ms) {
#pragma unroll
    for (int ns = 0; ns < 4; ++ns) {
      const int n = n0 + ns * 16 + (lane & 15);
      const float bv = bias[n];
#pragma unroll
      for (int r = 0; r < 8; ++r) {
        const int m = row0 + ms * 16 + r + ((lane >> 4) << 3);
        const float v = acc[ms][ns][r] + bv;
        if (OUT_F32) ((float*)outp)[(size_t)m * N + n] = v;
        else         ((__bf16*)outp)[(size_t)m * N + n] = f2bf(v);
      }
    }
  }
}

// ---------- attention: per (b,h), flash-style over 64-key tiles ----------
// 256 threads = 8 waves; block handles 128 q rows (16 per wave), so each
// staged K/V tile is amortized over 2x more WMMA work.

#define LDT 72 // padded LDS row (multiple of 8 elems; 144B rows keep 16B align)

__global__ __launch_bounds__(256)
void attention(const __bf16* __restrict__ Q, const __bf16* __restrict__ Kmat,
               const __bf16* __restrict__ V, __bf16* __restrict__ O) {
  __shared__ __align__(16) __bf16 Kt[64 * LDT];      // [key][d]
  __shared__ __align__(16) __bf16 Vt[64 * LDT];      // [d][key] (transposed)
  __shared__ __align__(16) __bf16 Pb[8 * 16 * LDT];  // per-wave P tile [16][64]

  const int b = blockIdx.y >> 4;
  const int h = blockIdx.y & 15;
  const int q0 = blockIdx.x * 128;
  const int wave = threadIdx.x >> 5;
  const int lane = threadIdx.x & 31;
  const int tid = threadIdx.x;

  const size_t hb = (size_t)b * SEQ * D_MODEL + (size_t)h * D_K;

  // Q fragments for this wave's 16 rows (held in registers for whole kernel)
  const __bf16* Qw = Q + hb + (size_t)(q0 + wave * 16) * D_MODEL;
  const v16bf aq0 = load_fragA(Qw, D_MODEL);
  const v16bf aq1 = load_fragA(Qw + 32, D_MODEL);

  const v8f zero8 = {0.f, 0.f, 0.f, 0.f, 0.f, 0.f, 0.f, 0.f};
  v8f o[4] = {zero8, zero8, zero8, zero8};
  float mi[8], li[8];
#pragma unroll
  for (int r = 0; r < 8; ++r) { mi[r] = -1.0e30f; li[r] = 0.f; }

  const float sc = 0.125f * 1.44269504088896340736f; // log2(e)/sqrt(d_k)

  for (int kt = 0; kt < SEQ; kt += 64) {
    // ---- stage K tile (async global->LDS when available) and V^T tile ----
    const __bf16* Kg = Kmat + hb + (size_t)kt * D_MODEL;
    const __bf16* Vg = V    + hb + (size_t)kt * D_MODEL;
    for (int c = tid; c < 512; c += 256) { // 512 chunks of 8 bf16 (16B)
      const int key = c >> 3, d8 = (c & 7) << 3;
#if USE_ASYNC_LDS
      __builtin_amdgcn_global_load_async_to_lds_b128(
          (AS1 v4i*)(void*)(Kg + (size_t)key * D_MODEL + d8),
          (AS3 v4i*)(void*)&Kt[key * LDT + d8], 0, 0);
#else
      *(bf16x8*)&Kt[key * LDT + d8] =
          *(const bf16x8*)(Kg + (size_t)key * D_MODEL + d8);
#endif
      const bf16x8 vv = *(const bf16x8*)(Vg + (size_t)key * D_MODEL + d8);
#pragma unroll
      for (int j = 0; j < 8; ++j) Vt[(d8 + j) * LDT + key] = vv[j];
    }
#if USE_ASYNC_LDS
    wait_async0();
#endif
    __syncthreads();

    // ---- scores S = Q @ K^T : 16x64 per wave, 8 WMMAs ----
    v8f s[4] = {zero8, zero8, zero8, zero8};
#pragma unroll
    for (int g = 0; g < 2; ++g) {
      const v16bf a = g ? aq1 : aq0;
#pragma unroll
      for (int ns = 0; ns < 4; ++ns) {
        const v16bf bk = load_fragB(&Kt[ns * 16 * LDT + g * 32], LDT);
        s[ns] = __builtin_amdgcn_wmma_f32_16x16x32_bf16(
            false, a, false, bk, (short)0, s[ns], false, false);
      }
    }

    // ---- online softmax (exp2 domain), P written to LDS as bf16 ----
#pragma unroll
    for (int r = 0; r < 8; ++r) {
      float t0 = s[0][r] * sc, t1 = s[1][r] * sc;
      float t2 = s[2][r] * sc, t3 = s[3][r] * sc;
      float rm = fmaxf(fmaxf(t0, t1), fmaxf(t2, t3));
      rm = fmaxf(rm, __shfl_xor(rm, 1, 32));
      rm = fmaxf(rm, __shfl_xor(rm, 2, 32));
      rm = fmaxf(rm, __shfl_xor(rm, 4, 32));
      rm = fmaxf(rm, __shfl_xor(rm, 8, 32));
      const float mnew = fmaxf(mi[r], rm);
      const float corr = exp2f(mi[r] - mnew);
      const float p0 = exp2f(t0 - mnew), p1 = exp2f(t1 - mnew);
      const float p2 = exp2f(t2 - mnew), p3 = exp2f(t3 - mnew);
      float rs = p0 + p1 + p2 + p3;
      rs += __shfl_xor(rs, 1, 32);
      rs += __shfl_xor(rs, 2, 32);
      rs += __shfl_xor(rs, 4, 32);
      rs += __shfl_xor(rs, 8, 32);
      li[r] = li[r] * corr + rs;
      mi[r] = mnew;
      o[0][r] *= corr; o[1][r] *= corr; o[2][r] *= corr; o[3][r] *= corr;
      const int m = r + ((lane >> 4) << 3);
      __bf16* pp = &Pb[(wave * 16 + m) * LDT + (lane & 15)];
      pp[0]  = f2bf(p0);
      pp[16] = f2bf(p1);
      pp[32] = f2bf(p2);
      pp[48] = f2bf(p3);
    }
    // same-wave LDS store->load is in-order; no barrier needed for Pb

    // ---- O += P @ V : 8 WMMAs (B fragments from transposed V tile) ----
#pragma unroll
    for (int g = 0; g < 2; ++g) {
      const v16bf ap = load_fragA(&Pb[wave * 16 * LDT + g * 32], LDT);
#pragma unroll
      for (int ns = 0; ns < 4; ++ns) {
        const v16bf bv = load_fragB(&Vt[ns * 16 * LDT + g * 32], LDT);
        o[ns] = __builtin_amdgcn_wmma_f32_16x16x32_bf16(
            false, ap, false, bv, (short)0, o[ns], false, false);
      }
    }
    __syncthreads(); // protect Kt/Vt before next tile overwrites
  }

  // ---- normalize and store attention output (bf16, [B,S,D_MODEL]) ----
#pragma unroll
  for (int ns = 0; ns < 4; ++ns) {
    const int n = ns * 16 + (lane & 15);
#pragma unroll
    for (int r = 0; r < 8; ++r) {
      const int m = r + ((lane >> 4) << 3);
      O[hb + (size_t)(q0 + wave * 16 + m) * D_MODEL + n] =
          f2bf(o[ns][r] * (1.0f / li[r]));
    }
  }
}

// ---------- launcher ----------

extern "C" void kernel_launch(void* const* d_in, const int* in_sizes, int n_in,
                              void* d_out, int out_size, void* d_ws, size_t ws_size,
                              hipStream_t stream) {
  const float* x  = (const float*)d_in[0];
  const float* wq = (const float*)d_in[1];
  const float* bq = (const float*)d_in[2];
  const float* wk = (const float*)d_in[3];
  const float* bk = (const float*)d_in[4];
  const float* wv = (const float*)d_in[5];
  const float* bv = (const float*)d_in[6];
  const float* wo = (const float*)d_in[7];
  const float* bo = (const float*)d_in[8];

  const size_t NX = (size_t)M_TOTAL * D_MODEL;  // 8M elems
  const size_t NW = (size_t)D_MODEL * D_MODEL;  // 1M elems

  char* ws = (char*)d_ws;
  __bf16* xh  = (__bf16*)ws;  ws += NX * 2;
  __bf16* wqh = (__bf16*)ws;  ws += NW * 2;
  __bf16* wkh = (__bf16*)ws;  ws += NW * 2;
  __bf16* wvh = (__bf16*)ws;  ws += NW * 2;
  __bf16* woh = (__bf16*)ws;  ws += NW * 2;
  __bf16* Qh  = (__bf16*)ws;  ws += NX * 2;
  __bf16* Kh  = (__bf16*)ws;  ws += NX * 2;
  __bf16* Vh  = (__bf16*)ws;  ws += NX * 2;
  __bf16* Oh  = (__bf16*)ws;  ws += NX * 2;

  // fp32 -> bf16 (8 elems per thread-iteration)
  cvt_f32_bf16<<<1024, 256, 0, stream>>>(x,  xh,  (int)(NX / 8));
  cvt_f32_bf16<<<256, 256, 0, stream>>>(wq, wqh, (int)(NW / 8));
  cvt_f32_bf16<<<256, 256, 0, stream>>>(wk, wkh, (int)(NW / 8));
  cvt_f32_bf16<<<256, 256, 0, stream>>>(wv, wvh, (int)(NW / 8));
  cvt_f32_bf16<<<256, 256, 0, stream>>>(wo, woh, (int)(NW / 8));

  // QKV projections: (8192/32)*(1024/64) = 4096 waves -> 512 blocks x 8 waves
  gemm_xwT<false><<<512, 256, 0, stream>>>(xh, wqh, bq, (void*)Qh, M_TOTAL);
  gemm_xwT<false><<<512, 256, 0, stream>>>(xh, wkh, bk, (void*)Kh, M_TOTAL);
  gemm_xwT<false><<<512, 256, 0, stream>>>(xh, wvh, bv, (void*)Vh, M_TOTAL);

  // attention: (S/128) x (B*H) blocks of 256 threads
  attention<<<dim3(SEQ / 128, BATCH * N_HEADS), 256, 0, stream>>>(Qh, Kh, Vh, Oh);

  // output projection -> fp32 d_out
  gemm_xwT<true><<<512, 256, 0, stream>>>(Oh, woh, bo, d_out, M_TOTAL);
}
